// SpikingResNet_9242769621463
// MI455X (gfx1250) — matrix operand order repaired and checked by reading, not compiled, not run on hardware
//
#include <hip/hip_runtime.h>
#include <hip/hip_bf16.h>
#include <stdint.h>

typedef __attribute__((ext_vector_type(16))) _Float16 v16h;
typedef __attribute__((ext_vector_type(8)))  float    v8f;

#define DECAY  0.8f
#define THRESH 0.5f

// Padded activation layout: [64 ch][30 rows][32 cols] halfs, interior at (+1,+1)
#define PROW   32
#define PCH    960            // 30*32
#define PIMG   ((size_t)64 * PCH)   // per (b) image, halfs

// ---------------------------------------------------------------------------
// Async weight-tile staging: 32 lanes x 16B per issue (CDNA5 async-to-LDS).
// ---------------------------------------------------------------------------
__device__ __forceinline__ void async_copy_b128(unsigned lds_off, const void* gaddr) {
  asm volatile("global_load_async_to_lds_b128 %0, %1, off"
               :: "v"(lds_off), "v"(gaddr) : "memory");
}
__device__ __forceinline__ void wait_async0() {
  asm volatile("s_wait_asynccnt 0" ::: "memory");
}

// ---------------------------------------------------------------------------
// WMMA fragment loaders (contiguous k order: lane half h holds k 16h..16h+15, wave32).
// A tile in LDS: row-major [64 m][32 k]. B tile in LDS: n-major [64 n][32 k].
// Both give each lane contiguous 16B runs -> ds_load_b128.
// ---------------------------------------------------------------------------
__device__ __forceinline__ v16h frag_a(const _Float16* tile /* + m0*32 */, int lane) {
  int row  = lane & 15;
  int kofs = (lane & 16) ? 16 : 0;
  const _Float16* p = tile + row * 32 + kofs;
  v16h a;
#pragma unroll
  for (int h = 0; h < 8; ++h) a[h] = p[h];        // K kofs..kofs+7
#pragma unroll
  for (int h = 0; h < 8; ++h) a[8 + h] = p[8 + h]; // K kofs+8..kofs+15
  return a;
}

__device__ __forceinline__ v16h frag_b(const _Float16* tile /* + n0*32 */, int lane) {
  int col  = lane & 15;
  int kofs = (lane & 16) ? 16 : 0;
  const _Float16* p = tile + col * 32 + kofs;
  v16h b;
#pragma unroll
  for (int h = 0; h < 16; ++h) b[h] = p[h];       // K kofs..kofs+15 contiguous
  return b;
}

// ---------------------------------------------------------------------------
// Weight conversions
// ---------------------------------------------------------------------------
__global__ void cvt_f32_f16(const float* __restrict__ in, _Float16* __restrict__ out, int n) {
  int i = blockIdx.x * 256 + threadIdx.x;
  if (i < n) out[i] = (_Float16)in[i];
}

// conv1 weights [64][980] f32 -> [64][992] f16, zero-padded K (so async-staged
// garbage never happens and padded K rows multiply against exact zeros).
__global__ void cvt_conv1_w(const float* __restrict__ in, _Float16* __restrict__ out) {
  int i = blockIdx.x * 256 + threadIdx.x;
  if (i >= 64 * 992) return;
  int row = i / 992, k = i - row * 992;
  out[i] = (k < 980) ? (_Float16)in[row * 980 + k] : (_Float16)0.0f;
}

// ---------------------------------------------------------------------------
// conv1: 7x7 s2 p3, implicit GEMM M=64, K=992(980), N=56x56.
// N tile = 4 rows x 16 cols; grid = (14*4, B); 128 threads (4 waves).
// Fused LIF epilogue.
// ---------------------------------------------------------------------------
__global__ __launch_bounds__(128)
void conv1_wmma(const float* __restrict__ x,      // [B][8][20][112][112]
                const _Float16* __restrict__ Wh,  // [64][992] padded
                float* __restrict__ c_mem,        // [B][64][3136]
                _Float16* __restrict__ spike,     // [B][64][3136] dense
                int t) {
  constexpr int KP = 992, KV = 980, HW = 3136;
  alignas(64) __shared__ _Float16 sA[64 * 32];
  alignas(64) __shared__ _Float16 sB[64 * 32];   // n-major [n][k]

  const int b    = blockIdx.y;
  const int oh0  = (blockIdx.x >> 2) * 4;
  const int ow0  = (blockIdx.x & 3) * 16;        // cols 48..63 partially dead
  const int tid  = threadIdx.x;
  const int lane = tid & 31;
  const int wave = tid >> 5;
  const int wm   = wave & 1;
  const int wn   = wave >> 1;

  const float* xb = x + (((size_t)b * 8 + t) * 20) * 112 * 112;

  // im2col coords for this thread's B slot
  const int nB  = tid >> 1;            // 0..63
  const int krb = (tid & 1) * 16;      // 0 or 16
  const int oh  = oh0 + (nB >> 4);
  const int ow  = ow0 + (nB & 15);

  v8f acc[2][2] = {};

  for (int k0 = 0; k0 < KP; k0 += 32) {
    // --- async-stage A: Wh[0:64][k0:k0+32], 256 slots of 8 halfs (16B) ---
#pragma unroll
    for (int j = 0; j < 2; ++j) {
      int s   = tid + j * 128;
      int row = s >> 2, kq = (s & 3) * 8;
      async_copy_b128((unsigned)(uintptr_t)&sA[row * 32 + kq],
                      Wh + (size_t)row * KP + k0 + kq);
    }
    // --- branchless im2col gather for B (clamped address + select) ---
    {
      _Float16 tmp[16];
#pragma unroll
      for (int j = 0; j < 16; ++j) {
        int kk  = k0 + krb + j;
        int ci  = kk / 49;
        int rem = kk - ci * 49;
        int fy  = rem / 7;
        int fx  = rem - fy * 7;
        int ih  = 2 * oh + fy - 3;
        int iw  = 2 * ow + fx - 3;
        bool ok = (kk < KV) & ((unsigned)ih < 112u) & ((unsigned)iw < 112u);
        int ihc = ih < 0 ? 0 : (ih > 111 ? 111 : ih);
        int iwc = iw < 0 ? 0 : (iw > 111 ? 111 : iw);
        int cic = ci > 19 ? 19 : ci;
        float vv = xb[((size_t)cic * 112 + ihc) * 112 + iwc];
        tmp[j] = ok ? (_Float16)vv : (_Float16)0.0f;
      }
      *(v16h*)&sB[nB * 32 + krb] = *(v16h*)tmp;
    }
    wait_async0();
    __syncthreads();
#pragma unroll
    for (int mi = 0; mi < 2; ++mi) {
      v16h a = frag_a(sA + (wm * 32 + mi * 16) * 32, lane);
#pragma unroll
      for (int ni = 0; ni < 2; ++ni) {
        v16h bb = frag_b(sB + (wn * 32 + ni * 16) * 32, lane);
        acc[mi][ni] = __builtin_amdgcn_wmma_f32_16x16x32_f16(
            false, a, false, bb, (short)0, acc[mi][ni], false, false);
      }
    }
    __syncthreads();
  }

  // Fused LIF epilogue (dense 56x56 out)
  float*    memB = c_mem + (size_t)b * 64 * HW;
  _Float16* spkB = spike + (size_t)b * 64 * HW;
  const int col  = lane & 15;
  const int rofs = (lane & 16) ? 8 : 0;
#pragma unroll
  for (int mi = 0; mi < 2; ++mi)
#pragma unroll
    for (int ni = 0; ni < 2; ++ni) {
      int nn  = wn * 32 + ni * 16 + col;
      int soh = oh0 + (nn >> 4);
      int sow = ow0 + (nn & 15);
      if (sow < 56) {
#pragma unroll
        for (int r = 0; r < 8; ++r) {
          int m = wm * 32 + mi * 16 + rofs + r;
          size_t off = (size_t)m * HW + soh * 56 + sow;
          float mv = memB[off] * DECAY + acc[mi][ni][r];
          memB[off] = mv;
          spkB[off] = (_Float16)((mv > THRESH) ? 1.0f : 0.0f);
        }
      }
    }
}

// ---------------------------------------------------------------------------
// 3x3 avg pool s2 p1 (count_include_pad): dense 56x56 -> padded 30x32 layout
// ---------------------------------------------------------------------------
__global__ void pool3x3s2(const _Float16* __restrict__ in,   // [B*64][56][56]
                          _Float16* __restrict__ out) {      // [B*64][30][32] padded
  int i = blockIdx.x * 256 + threadIdx.x;
  if (i >= 16 * 64 * 784) return;
  int p  = i % 784;
  int bm = i / 784;
  int oh = p / 28, ow = p % 28;
  const _Float16* src = in + (size_t)bm * 3136;
  float s = 0.0f;
#pragma unroll
  for (int u = 0; u < 3; ++u) {
    int ih = 2 * oh - 1 + u;
    if ((unsigned)ih >= 56u) continue;
#pragma unroll
    for (int v = 0; v < 3; ++v) {
      int iw = 2 * ow - 1 + v;
      if ((unsigned)iw >= 56u) continue;
      s += (float)src[ih * 56 + iw];
    }
  }
  out[(size_t)bm * PCH + (oh + 1) * PROW + (ow + 1)] = (_Float16)(s * (1.0f / 9.0f));
}

// ---------------------------------------------------------------------------
// Residual-block 3x3 conv, implicit GEMM M=64, K=576, N=28x28.
// Inputs/outputs in padded [64][30][32] layout -> zero bounds checks.
// N tile = 4 rows x 16 cols; grid = (7*2, B); 128 threads.
// ---------------------------------------------------------------------------
__global__ __launch_bounds__(128)
void block_conv_wmma(const _Float16* __restrict__ X,     // [B] padded
                     const _Float16* __restrict__ Wh,    // [64][576]
                     float* __restrict__ mem,            // [B][64][784] dense
                     const _Float16* __restrict__ resid, // [B] padded or null
                     _Float16* __restrict__ outSpike) {  // [B] padded
  constexpr int KTOT = 576, HW = 784;
  alignas(64) __shared__ _Float16 sA[64 * 32];
  alignas(64) __shared__ _Float16 sB[64 * 32];   // n-major [n][k]

  const int b    = blockIdx.y;
  const int oh0  = (blockIdx.x >> 1) * 4;
  const int ow0  = (blockIdx.x & 1) * 16;        // cols 28..31 dead
  const int tid  = threadIdx.x;
  const int lane = tid & 31;
  const int wave = tid >> 5;
  const int wm   = wave & 1;
  const int wn   = wave >> 1;

  const _Float16* Xb = X + (size_t)b * PIMG;

  const int nB  = tid >> 1;
  const int krb = (tid & 1) * 16;
  const int oh  = oh0 + (nB >> 4);
  const int ow  = ow0 + (nB & 15);   // 0..31; padded reads stay in-buffer

  v8f acc[2][2] = {};

  for (int k0 = 0; k0 < KTOT; k0 += 32) {
    // --- async-stage A: Wh[0:64][k0:k0+32] ---
#pragma unroll
    for (int j = 0; j < 2; ++j) {
      int s   = tid + j * 128;
      int row = s >> 2, kq = (s & 3) * 8;
      async_copy_b128((unsigned)(uintptr_t)&sA[row * 32 + kq],
                      Wh + (size_t)row * KTOT + k0 + kq);
    }
    // --- fully branchless padded im2col gather for B ---
    {
      _Float16 tmp[16];
#pragma unroll
      for (int j = 0; j < 16; ++j) {
        int kk  = k0 + krb + j;
        int ci  = kk / 9;
        int rem = kk - ci * 9;
        int dy  = rem / 3;             // 0..2 (padded row = oh + dy)
        int dx  = rem - dy * 3;        // 0..2 (padded col = ow + dx)
        tmp[j] = Xb[(size_t)ci * PCH + (oh + dy) * PROW + (ow + dx)];
      }
      *(v16h*)&sB[nB * 32 + krb] = *(v16h*)tmp;
    }
    wait_async0();
    __syncthreads();
#pragma unroll
    for (int mi = 0; mi < 2; ++mi) {
      v16h a = frag_a(sA + (wm * 32 + mi * 16) * 32, lane);
#pragma unroll
      for (int ni = 0; ni < 2; ++ni) {
        v16h bb = frag_b(sB + (wn * 32 + ni * 16) * 32, lane);
        acc[mi][ni] = __builtin_amdgcn_wmma_f32_16x16x32_f16(
            false, a, false, bb, (short)0, acc[mi][ni], false, false);
      }
    }
    __syncthreads();
  }

  float*          memB = mem + (size_t)b * 64 * HW;
  const _Float16* resB = resid ? resid + (size_t)b * PIMG : nullptr;
  _Float16*       outB = outSpike + (size_t)b * PIMG;
  const int col  = lane & 15;
  const int rofs = (lane & 16) ? 8 : 0;
#pragma unroll
  for (int mi = 0; mi < 2; ++mi)
#pragma unroll
    for (int ni = 0; ni < 2; ++ni) {
      int nn  = wn * 32 + ni * 16 + col;
      int soh = oh0 + (nn >> 4);
      int sow = ow0 + (nn & 15);
      if (sow < 28) {
#pragma unroll
        for (int r = 0; r < 8; ++r) {
          int m = wm * 32 + mi * 16 + rofs + r;
          size_t doff = (size_t)m * HW + soh * 28 + sow;           // dense mem
          size_t poff = (size_t)m * PCH + (soh + 1) * PROW + (sow + 1); // padded act
          float mv = memB[doff] * DECAY + acc[mi][ni][r];
          if (resB) mv += (float)resB[poff];
          memB[doff] = mv;
          outB[poff] = (_Float16)((mv > THRESH) ? 1.0f : 0.0f);
        }
      }
    }
}

// ---------------------------------------------------------------------------
// Head: concat(s2[0,2,4,6]) -> avgpool 7x7 -> [4096] -> FC 101 -> LIF
// ---------------------------------------------------------------------------
__global__ __launch_bounds__(256)
void fc_step(const _Float16* __restrict__ s2all,  // [8][B] padded images
             const _Float16* __restrict__ fcw,    // [101][4096]
             const float* __restrict__ fcb,
             float* __restrict__ fc_mem,          // [B][101]
             float* __restrict__ fc_sum) {        // [B][101]
  __shared__ _Float16 feat[4096];
  const int b = blockIdx.x;
  const size_t blkStride = (size_t)16 * PIMG;

  for (int idx = threadIdx.x; idx < 4096; idx += 256) {
    int cc = idx >> 4;
    int q  = idx & 15;
    int qh = q >> 2, qw = q & 3;
    int bi = (cc >> 6) * 2;            // blocks 0,2,4,6
    int m  = cc & 63;
    const _Float16* base = s2all + blkStride * bi + ((size_t)b * 64 + m) * PCH;
    float s = 0.0f;
#pragma unroll
    for (int u = 0; u < 7; ++u)
#pragma unroll
      for (int v = 0; v < 7; ++v)
        s += (float)base[(qh * 7 + u + 1) * PROW + (qw * 7 + v + 1)];
    feat[idx] = (_Float16)(s * (1.0f / 49.0f));
  }
  __syncthreads();

  for (int o = threadIdx.x; o < 101; o += 256) {
    float acc = fcb[o];
    const _Float16* wrow = fcw + (size_t)o * 4096;
    for (int k = 0; k < 4096; ++k)
      acc += (float)feat[k] * (float)wrow[k];
    float mv = fc_mem[b * 101 + o] * DECAY + acc;
    fc_mem[b * 101 + o] = mv;
    fc_sum[b * 101 + o] += (mv > THRESH) ? 1.0f : 0.0f;
  }
}

__global__ void finalize_div(const float* __restrict__ fc_sum, float* __restrict__ out, int n) {
  int i = blockIdx.x * 256 + threadIdx.x;
  if (i < n) out[i] = fc_sum[i] * (1.0f / 8.0f);
}

// ---------------------------------------------------------------------------
// Host launch sequence (graph-capture safe: kernels + hipMemsetAsync only)
// ---------------------------------------------------------------------------
extern "C" void kernel_launch(void* const* d_in, const int* in_sizes, int n_in,
                              void* d_out, int out_size, void* d_ws, size_t ws_size,
                              hipStream_t stream) {
  const float* x    = (const float*)d_in[0];
  const float* w1f  = (const float*)d_in[1];
  const float* wb1f = (const float*)d_in[2];
  const float* wb2f = (const float*)d_in[3];
  const float* wfcf = (const float*)d_in[4];
  const float* fcb  = (const float*)d_in[5];

  char* p = (char*)d_ws;
  auto carve = [&](size_t bytes) -> char* {
    char* r = p;
    p += (bytes + 255) & ~(size_t)255;
    return r;
  };

  const size_t ACT  = (size_t)16 * PIMG;          // padded activation tensor (halfs)
  const size_t MEMB = (size_t)16 * 64 * 784;      // dense block membrane (elems)

  _Float16* wc1    = (_Float16*)carve((size_t)64 * 992 * 2);
  _Float16* wb1    = (_Float16*)carve((size_t)294912 * 2);
  _Float16* wb2    = (_Float16*)carve((size_t)294912 * 2);
  _Float16* wfc    = (_Float16*)carve((size_t)413696 * 2);
  float*    c_mem  = (float*)carve((size_t)16 * 64 * 3136 * 4);
  _Float16* spike1 = (_Float16*)carve((size_t)16 * 64 * 3136 * 2);
  _Float16* cur    = (_Float16*)carve(ACT * 2 + 256);
  _Float16* s1     = (_Float16*)carve(ACT * 2 + 256);
  _Float16* s2     = (_Float16*)carve((size_t)8 * ACT * 2 + 256);
  float*    b1m    = (float*)carve((size_t)8 * MEMB * 4);
  float*    b2m    = (float*)carve((size_t)8 * MEMB * 4);
  float*    fc_mem = (float*)carve((size_t)1616 * 4);
  float*    fc_sum = (float*)carve((size_t)1616 * 4);

  // Weight conversions (deterministic every call)
  cvt_conv1_w<<<(64 * 992 + 255) / 256, 256, 0, stream>>>(w1f, wc1);
  cvt_f32_f16<<<(294912 + 255) / 256, 256, 0, stream>>>(wb1f, wb1, 294912);
  cvt_f32_f16<<<(294912 + 255) / 256, 256, 0, stream>>>(wb2f, wb2, 294912);
  cvt_f32_f16<<<(413696 + 255) / 256, 256, 0, stream>>>(wfcf, wfc, 413696);

  // Zero membrane states and padded-activation halos
  hipMemsetAsync(c_mem, 0, (size_t)16 * 64 * 3136 * 4, stream);
  hipMemsetAsync(b1m, 0, (size_t)8 * MEMB * 4, stream);
  hipMemsetAsync(b2m, 0, (size_t)8 * MEMB * 4, stream);
  hipMemsetAsync(cur, 0, ACT * 2 + 256, stream);
  hipMemsetAsync(s1, 0, ACT * 2 + 256, stream);
  hipMemsetAsync(s2, 0, (size_t)8 * ACT * 2 + 256, stream);
  hipMemsetAsync(fc_mem, 0, (size_t)1616 * 4, stream);
  hipMemsetAsync(fc_sum, 0, (size_t)1616 * 4, stream);

  for (int t = 0; t < 8; ++t) {
    conv1_wmma<<<dim3(56, 16), 128, 0, stream>>>(x, wc1, c_mem, spike1, t);
    pool3x3s2<<<(16 * 64 * 784 + 255) / 256, 256, 0, stream>>>(spike1, cur);

    const _Float16* curp = cur;
    for (int i = 0; i < 8; ++i) {
      block_conv_wmma<<<dim3(14, 16), 128, 0, stream>>>(
          curp, wb1 + (size_t)i * 64 * 576, b1m + (size_t)i * MEMB,
          (const _Float16*)nullptr, s1);
      block_conv_wmma<<<dim3(14, 16), 128, 0, stream>>>(
          s1, wb2 + (size_t)i * 64 * 576, b2m + (size_t)i * MEMB,
          curp, s2 + (size_t)i * ACT);
      curp = s2 + (size_t)i * ACT;
    }
    fc_step<<<16, 256, 0, stream>>>(s2, wfc, fcb, fc_mem, fc_sum);
  }
  finalize_div<<<(1616 + 255) / 256, 256, 0, stream>>>(fc_sum, (float*)d_out, 1616);
}
